// BrainNATLayer_35072702939329
// MI455X (gfx1250) — matrix operand order repaired and checked by reading, not compiled, not run on hardware
//
#include <hip/hip_runtime.h>
#include <hip/hip_bf16.h>

typedef __bf16 bf16_t;
typedef __bf16 bf16x8  __attribute__((ext_vector_type(8)));
typedef __bf16 bf16x16 __attribute__((ext_vector_type(16)));
typedef float  f32x8   __attribute__((ext_vector_type(8)));
typedef unsigned int u32x4 __attribute__((ext_vector_type(4)));
typedef int          i32x4 __attribute__((ext_vector_type(4)));
typedef int          i32x8 __attribute__((ext_vector_type(8)));

#define BB   2
#define NN   2048
#define DD   768
#define HH   12
#define HDD  64
#define KW   32
#define RR   256
#define NM   1792          // NN - RR
#define MLPH 6144
#define KCHUNK 128         // K-chunk staged per TDM transfer (all K % 128 == 0)

#if defined(__HIP_DEVICE_COMPILE__) && __has_builtin(__builtin_amdgcn_tensor_load_to_lds) && __has_builtin(__builtin_amdgcn_s_wait_tensorcnt)
#define HAVE_TDM 1
#else
#define HAVE_TDM 0
#endif

// ---------------------------------------------------------------------------
// f32 -> bf16 conversion (weights, once per launch)
// ---------------------------------------------------------------------------
__global__ void cvt_bf16_kernel(const float* __restrict__ src, bf16_t* __restrict__ dst, int n) {
  int i = blockIdx.x * 256 + threadIdx.x;
  if (i < n) dst[i] = (bf16_t)src[i];
}

// ---------------------------------------------------------------------------
// LayerNorm over D=768, one block (256 thr) per row, bf16 output
// ---------------------------------------------------------------------------
__global__ __launch_bounds__(256)
void ln_kernel(const float* __restrict__ x, const float* __restrict__ w,
               const float* __restrict__ b, bf16_t* __restrict__ out) {
  int row = blockIdx.x;
  int t   = threadIdx.x;
  const float* xr = x + (size_t)row * DD;
  float v0 = xr[t], v1 = xr[t + 256], v2 = xr[t + 512];
  __shared__ float r1[256], r2[256];
  r1[t] = v0 + v1 + v2;
  r2[t] = v0 * v0 + v1 * v1 + v2 * v2;
  __syncthreads();
  for (int o = 128; o > 0; o >>= 1) {
    if (t < o) { r1[t] += r1[t + o]; r2[t] += r2[t + o]; }
    __syncthreads();
  }
  float mean = r1[0] * (1.f / DD);
  float var  = r2[0] * (1.f / DD) - mean * mean;
  float rs   = rsqrtf(var + 1e-5f);
  bf16_t* orow = out + (size_t)row * DD;
#pragma unroll
  for (int e = 0; e < 3; e++) {
    int c = t + 256 * e;
    orow[c] = (bf16_t)((xr[c] - mean) * rs * w[c] + b[c]);
  }
}

// ---------------------------------------------------------------------------
// TDM issue helper: DMA a 64 x KCHUNK bf16 weight tile (rows = N tile at
// colBase, cols = K window at kc) into LDS at ldsOff.
// D# per CDNA5 ISA (2-D tensor, groups 2/3 zero).
// ---------------------------------------------------------------------------
#if HAVE_TDM
__device__ __forceinline__ void tdm_issue_b_tile(const bf16_t* W, int colBase, int kc,
                                                 int Kk, int Nn, unsigned ldsOff) {
  unsigned long long ga =
      (unsigned long long)(size_t)(const void*)W + ((size_t)colBase * Kk + kc) * 2ull;
  u32x4 g0;
  g0[0] = 1u;                                            // count=1, no gather
  g0[1] = ldsOff;                                        // lds_addr
  g0[2] = (unsigned)ga;                                  // global_addr[31:0]
  g0[3] = (unsigned)((ga >> 32) & 0x01FFFFFFu) | (2u << 30); // addr[56:32] | type=2
  i32x8 g1;
  g1[0] = (int)(1u << 16);                               // data_size = 2 bytes
  g1[1] = (int)(((unsigned)Kk & 0xFFFFu) << 16);         // tensor_dim0[15:0]
  g1[2] = (int)(((unsigned)Kk >> 16) | (((unsigned)Nn & 0xFFFFu) << 16));
  g1[3] = (int)(((unsigned)Nn >> 16) | ((unsigned)KCHUNK << 16)); // tile_dim0
  g1[4] = 64;                                            // tile_dim1 = 64 rows
  g1[5] = (int)(unsigned)Kk;                             // tensor_dim0_stride lo
  g1[6] = 0;
  g1[7] = 0;
  i32x4 gz = {0, 0, 0, 0};
#if __clang_major__ >= 23
  i32x8 z8 = {0, 0, 0, 0, 0, 0, 0, 0};
  __builtin_amdgcn_tensor_load_to_lds(g0, g1, gz, gz, z8, 0);
#else
  __builtin_amdgcn_tensor_load_to_lds(g0, g1, gz, gz, 0);
#endif
}
#endif

// ---------------------------------------------------------------------------
// bf16 WMMA GEMM: C[M,N] = A[M,K] * W[N,K]^T + bias, optional epilogue
//   out = resid + gamma * C   (resid/gamma may be null)
// One wave -> 16x64 tile, 4 waves/block -> 64x64 tile.
// Weight tiles are staged into LDS by the Tensor Data Mover with DOUBLE
// BUFFERING: wave 0 issues chunk c+1's TENSOR_LOAD_TO_LDS before the block
// computes chunk c, so the DMA overlaps the WMMA stream; the TENSORcnt wait
// lands after compute. The kk loop is fully unrolled (16 WMMAs straight-line)
// so the scheduler can hoist A/B loads across k-steps.
// Falls back to direct global loads without the builtin.
// ---------------------------------------------------------------------------
__global__ __launch_bounds__(128)
void gemm_bf16_kernel(const bf16_t* __restrict__ A, const bf16_t* __restrict__ W,
                      const float* __restrict__ bias,
                      float* __restrict__ outF, bf16_t* __restrict__ outB,
                      const float* __restrict__ resid, const float* __restrict__ gamma,
                      int M, int Nn, int Kk) {
  int lane    = threadIdx.x & 31;
  int wave    = threadIdx.x >> 5;
  int rowBase = blockIdx.y * 64 + wave * 16;
  int colBase = blockIdx.x * 64;

  int aRow  = rowBase + (lane & 15);
  int aKoff = (lane & 16) ? 16 : 0;    // contiguous k order: hi lanes hold K+16 block
  int bN    = lane & 15;
  int bKoff = (lane & 16) ? 16 : 0;    // contiguous k order: hi lanes hold K+16 block

  f32x8 acc[4];
#pragma unroll
  for (int j = 0; j < 4; j++) acc[j] = (f32x8){0.f,0.f,0.f,0.f,0.f,0.f,0.f,0.f};

  const bf16_t* aPtr = A + (size_t)aRow * Kk + aKoff;

#if HAVE_TDM
  __shared__ bf16_t Bs[2][64 * KCHUNK];                  // 2 x 16 KB double buffer
  unsigned ldsOff[2];
  ldsOff[0] = (unsigned)(size_t)(void*)&Bs[0][0];        // LDS aperture low 32 bits
  ldsOff[1] = (unsigned)(size_t)(void*)&Bs[1][0];

  int nch = Kk / KCHUNK;
  if (threadIdx.x < 32) {              // TDM is wave-level (EXEC ignored): wave 0 only
    tdm_issue_b_tile(W, colBase, 0, Kk, Nn, ldsOff[0]);
    __builtin_amdgcn_s_wait_tensorcnt(0);
  }
  __syncthreads();                     // buffer 0 ready

  for (int c = 0; c < nch; c++) {
    int kc = c * KCHUNK;
    if (c + 1 < nch && threadIdx.x < 32)                 // overlap next DMA w/ compute
      tdm_issue_b_tile(W, colBase, kc + KCHUNK, Kk, Nn, ldsOff[(c + 1) & 1]);

    __builtin_prefetch(aPtr + kc + KCHUNK, 0, 3);        // next A chunk, near cache

    const bf16_t* bBuf = &Bs[c & 1][0];
#pragma unroll
    for (int kk = 0; kk < KCHUNK; kk += 32) {            // straight-line 16 WMMAs
      bf16x8 alo = *(const bf16x8*)(aPtr + kc + kk);
      bf16x8 ahi = *(const bf16x8*)(aPtr + kc + kk + 8);
      bf16x16 af = __builtin_shufflevector(alo, ahi, 0,1,2,3,4,5,6,7,8,9,10,11,12,13,14,15);
      bf16x16 bfr[4];
#pragma unroll
      for (int j = 0; j < 4; j++)                        // 8 ds_load_b128 in flight
        bfr[j] = *(const bf16x16*)(bBuf + (size_t)(bN + j * 16) * KCHUNK + bKoff + kk);
#pragma unroll
      for (int j = 0; j < 4; j++)
        acc[j] = __builtin_amdgcn_wmma_f32_16x16x32_bf16(
            false, af, false, bfr[j], (short)0, acc[j], false, false);
    }
    __syncthreads();                   // all reads of Bs[c&1] done
    if (c + 1 < nch) {
      if (threadIdx.x < 32) __builtin_amdgcn_s_wait_tensorcnt(0);
      __syncthreads();                 // publish Bs[(c+1)&1]
    }
  }
#else
  for (int k0 = 0; k0 < Kk; k0 += 32) {
    bf16x8 alo = *(const bf16x8*)(aPtr + k0);
    bf16x8 ahi = *(const bf16x8*)(aPtr + k0 + 8);
    bf16x16 af = __builtin_shufflevector(alo, ahi, 0,1,2,3,4,5,6,7,8,9,10,11,12,13,14,15);
#pragma unroll
    for (int j = 0; j < 4; j++) {
      bf16x16 bfr = *(const bf16x16*)(W + (size_t)(colBase + bN + j * 16) * Kk + bKoff + k0);
      acc[j] = __builtin_amdgcn_wmma_f32_16x16x32_bf16(
          false, af, false, bfr, (short)0, acc[j], false, false);
    }
  }
#endif

  int rOff = (lane & 16) ? 8 : 0;
#pragma unroll
  for (int j = 0; j < 4; j++) {
    int col  = colBase + j * 16 + (lane & 15);
    float bv = bias  ? bias[col]  : 0.f;
    float gv = gamma ? gamma[col] : 1.f;
#pragma unroll
    for (int v = 0; v < 8; v++) {
      int row = rowBase + rOff + v;
      size_t oi = (size_t)row * Nn + col;
      float val = acc[j][v] + bv;
      if (resid) val = resid[oi] + gv * val;
      if (outF) outF[oi] = val;
      else      outB[oi] = (bf16_t)val;
    }
  }
}

// ---------------------------------------------------------------------------
// Local-window attention, K=32 window == wave32: lane == window slot.
// qkv layout: (B, N, 3*D) with channel = t*768 + h*64 + d.  Output bf16 (B,N,D).
// ---------------------------------------------------------------------------
__global__ __launch_bounds__(128)
void attn_kernel(const float* __restrict__ qkv, const unsigned char* __restrict__ mask,
                 bf16_t* __restrict__ attn_out) {
  int lane = threadIdx.x & 31;
  int wid  = blockIdx.x * 4 + (threadIdx.x >> 5);
  int n = wid % NN;
  int h = (wid / NN) % HH;
  int b = wid / (NN * HH);

  int idx = n + lane - KW / 2;
  idx = idx < 0 ? 0 : (idx > NN - 1 ? NN - 1 : idx);

  const float* qp = qkv + ((size_t)(b * NN + n))   * 2304 + h * HDD;
  const float* kp = qkv + ((size_t)(b * NN + idx)) * 2304 + 768 + h * HDD;
  float s = 0.f;
#pragma unroll 8
  for (int d = 0; d < HDD; d++) s += qp[d] * kp[d];
  s *= 0.125f;                              // HD^-0.5
  if (!mask[b * NN + idx]) s = -1e9f;

  float mx = s;
  for (int o = 16; o > 0; o >>= 1) mx = fmaxf(mx, __shfl_xor(mx, o));
  float e = __expf(s - mx);
  float sum = e;
  for (int o = 16; o > 0; o >>= 1) sum += __shfl_xor(sum, o);
  float a = e / sum;

  float o0 = 0.f, o1 = 0.f;
  int d0 = 2 * lane;
  for (int kk = 0; kk < KW; kk++) {
    float aw = __shfl(a, kk);
    int   ii = __shfl(idx, kk);
    const float* vp = qkv + ((size_t)(b * NN + ii)) * 2304 + 1536 + h * HDD + d0;
    o0 += aw * vp[0];
    o1 += aw * vp[1];
  }
  bf16_t* op = attn_out + ((size_t)(b * NN + n)) * DD + h * HDD + d0;
  op[0] = (bf16_t)o0;
  op[1] = (bf16_t)o1;
}

// ---------------------------------------------------------------------------
// ToMe metric: mean of k over heads, L2-normalized -> m (B, N, 64)
// ---------------------------------------------------------------------------
__global__ __launch_bounds__(128)
void metric_kernel(const float* __restrict__ qkv, float* __restrict__ m) {
  int lane = threadIdx.x & 31;
  int wid  = blockIdx.x * 4 + (threadIdx.x >> 5);
  int n = wid % NN;
  int b = wid / NN;
  int d0 = 2 * lane;
  const float* base = qkv + ((size_t)(b * NN + n)) * 2304 + 768 + d0;
  float s0 = 0.f, s1 = 0.f;
#pragma unroll
  for (int h = 0; h < HH; h++) { s0 += base[h * HDD]; s1 += base[h * HDD + 1]; }
  s0 *= (1.f / HH); s1 *= (1.f / HH);
  float q = s0 * s0 + s1 * s1;
  for (int o = 16; o > 0; o >>= 1) q += __shfl_xor(q, o);
  float inv = 1.f / (sqrtf(q) + 1e-6f);
  float* mo = m + ((size_t)(b * NN + n)) * HDD + d0;
  mo[0] = s0 * inv;
  mo[1] = s1 * inv;
}

// ---------------------------------------------------------------------------
// sim = a @ b^T, row-wise max+argmax.  a = m[:, ::2], b = m[:, 1::2].
// One wave per row i; lanes stride over j.
// ---------------------------------------------------------------------------
__global__ __launch_bounds__(256)
void sim_kernel(const float* __restrict__ m, float* __restrict__ nodeMax,
                int* __restrict__ nodeIdx) {
  __shared__ float aRow[8][64];
  int lane = threadIdx.x & 31;
  int wave = threadIdx.x >> 5;
  int wid  = blockIdx.x * 8 + wave;
  int b = wid >> 10;
  int i = wid & 1023;
  const float* ap = m + ((size_t)(b * NN + 2 * i)) * HDD;
  aRow[wave][2 * lane]     = ap[2 * lane];
  aRow[wave][2 * lane + 1] = ap[2 * lane + 1];
  __syncthreads();

  float best = -1e30f;
  int bestj = 0;
  for (int j = lane; j < 1024; j += 32) {
    const float* bp = m + ((size_t)(b * NN + 2 * j + 1)) * HDD;
    float dot = 0.f;
#pragma unroll 8
    for (int d = 0; d < 64; d++) dot += aRow[wave][d] * bp[d];
    if (dot > best) { best = dot; bestj = j; }
  }
  for (int o = 16; o > 0; o >>= 1) {
    float ov = __shfl_xor(best, o);
    int   oj = __shfl_xor(bestj, o);
    if (ov > best || (ov == best && oj < bestj)) { best = ov; bestj = oj; }
  }
  if (lane == 0) { nodeMax[b * 1024 + i] = best; nodeIdx[b * 1024 + i] = bestj; }
}

// ---------------------------------------------------------------------------
// Bitonic argsort of 1024 keys (descending, index tie-break) per batch in LDS.
// order = argsort(-nodeMax); dstIdx[r] = nodeIdx[order[r]] for r < 256.
// ---------------------------------------------------------------------------
__global__ __launch_bounds__(1024)
void sort_kernel(const float* __restrict__ nodeMax, const int* __restrict__ nodeIdx,
                 int* __restrict__ order, int* __restrict__ dstIdx) {
  __shared__ float sk[1024];
  __shared__ int   sv[1024];
  int b = blockIdx.x;
  int t = threadIdx.x;
  sk[t] = nodeMax[b * 1024 + t];
  sv[t] = t;
  __syncthreads();
  for (int size = 2; size <= 1024; size <<= 1) {
    for (int stride = size >> 1; stride > 0; stride >>= 1) {
      int ixj = t ^ stride;
      if (ixj > t) {
        float ka = sk[t], kb = sk[ixj];
        int   va = sv[t], vb = sv[ixj];
        bool aFirst = (ka > kb) || (ka == kb && va < vb);  // descending, stable
        bool desc   = ((t & size) == 0);
        if (desc ? !aFirst : aFirst) {
          sk[t] = kb; sk[ixj] = ka;
          sv[t] = vb; sv[ixj] = va;
        }
      }
      __syncthreads();
    }
  }
  order[b * 1024 + t] = sv[t];
  if (t < RR) dstIdx[b * RR + t] = nodeIdx[b * 1024 + sv[t]];
}

// ---------------------------------------------------------------------------
// Token merge: init dst accumulation with xb + cnt=1, scatter-add src rows,
// then emit merged x (B, 1792, 768).
// ---------------------------------------------------------------------------
__global__ __launch_bounds__(256)
void merge_init_kernel(const float* __restrict__ x1, float* __restrict__ dstbuf,
                       float* __restrict__ cnt) {
  int b = blockIdx.x >> 10;
  int j = blockIdx.x & 1023;
  const float* src = x1 + ((size_t)(b * NN + 2 * j + 1)) * DD;
  float* dst = dstbuf + ((size_t)(b * 1024 + j)) * DD;
#pragma unroll
  for (int e = 0; e < 3; e++) { int c = threadIdx.x + 256 * e; dst[c] = src[c]; }
  if (threadIdx.x == 0) cnt[b * 1024 + j] = 1.f;
}

__global__ __launch_bounds__(256)
void merge_scatter_kernel(const float* __restrict__ x1, const int* __restrict__ order,
                          const int* __restrict__ dstIdx, float* __restrict__ dstbuf,
                          float* __restrict__ cnt) {
  int b = blockIdx.x / RR;
  int r = blockIdx.x % RR;
  int s = order[b * 1024 + r];   // src_idx (xa row index)
  int d = dstIdx[b * RR + r];
  const float* src = x1 + ((size_t)(b * NN + 2 * s)) * DD;
  float* dst = dstbuf + ((size_t)(b * 1024 + d)) * DD;
  if (threadIdx.x == 0) atomicAdd(&cnt[b * 1024 + d], 1.f);
#pragma unroll
  for (int e = 0; e < 3; e++) {
    int c = threadIdx.x + 256 * e;
    atomicAdd(&dst[c], src[c]);
  }
}

__global__ __launch_bounds__(256)
void merge_final_kernel(const float* __restrict__ x1, const int* __restrict__ order,
                        const float* __restrict__ dstbuf, const float* __restrict__ cnt,
                        float* __restrict__ xm) {
  int b = blockIdx.x / NM;
  int i = blockIdx.x % NM;
  float* out = xm + ((size_t)(b * NM + i)) * DD;
  if (i < 1024 - RR) {                           // unmerged tokens from xa
    int u = order[b * 1024 + RR + i];
    const float* src = x1 + ((size_t)(b * NN + 2 * u)) * DD;
#pragma unroll
    for (int e = 0; e < 3; e++) { int c = threadIdx.x + 256 * e; out[c] = src[c]; }
  } else {                                       // merged dst tokens
    int j = i - (1024 - RR);
    float inv = 1.f / cnt[b * 1024 + j];
    const float* src = dstbuf + ((size_t)(b * 1024 + j)) * DD;
#pragma unroll
    for (int e = 0; e < 3; e++) { int c = threadIdx.x + 256 * e; out[c] = src[c] * inv; }
  }
}

// ---------------------------------------------------------------------------
// SwiGLU: g = silu(u[:, :3072]) * u[:, 3072:]  (bf16 in/out, f32 math)
// ---------------------------------------------------------------------------
__global__ __launch_bounds__(256)
void swiglu_kernel(const bf16_t* __restrict__ u, bf16_t* __restrict__ g) {
  size_t row = blockIdx.x;
  const bf16_t* ur = u + row * MLPH;
  bf16_t* gr = g + row * (MLPH / 2);
#pragma unroll
  for (int e = 0; e < 12; e++) {
    int c = threadIdx.x + 256 * e;
    float u1 = (float)ur[c];
    float u2 = (float)ur[c + MLPH / 2];
    float s  = u1 / (1.f + __expf(-u1));
    gr[c] = (bf16_t)(s * u2);
  }
}

// ---------------------------------------------------------------------------
// Host launcher
// ---------------------------------------------------------------------------
extern "C" void kernel_launch(void* const* d_in, const int* in_sizes, int n_in,
                              void* d_out, int out_size, void* d_ws, size_t ws_size,
                              hipStream_t stream) {
  (void)in_sizes; (void)n_in; (void)out_size; (void)ws_size;

  const float* x      = (const float*)d_in[0];
  const unsigned char* mask = (const unsigned char*)d_in[1];
  const float* w_qkv  = (const float*)d_in[2];
  const float* b_qkv  = (const float*)d_in[3];
  const float* w_proj = (const float*)d_in[4];
  const float* b_proj = (const float*)d_in[5];
  const float* n1w    = (const float*)d_in[6];
  const float* n1b    = (const float*)d_in[7];
  const float* n2w    = (const float*)d_in[8];
  const float* n2b    = (const float*)d_in[9];
  const float* g1     = (const float*)d_in[10];
  const float* g2     = (const float*)d_in[11];
  const float* fc1w   = (const float*)d_in[12];
  const float* fc1b   = (const float*)d_in[13];
  const float* fc2w   = (const float*)d_in[14];
  const float* fc2b   = (const float*)d_in[15];
  float* out = (float*)d_out;

  // workspace carve-up
  char* base = (char*)d_ws;
  size_t off = 0;
  auto alloc = [&](size_t bytes) -> void* {
    void* p = base + off;
    off += (bytes + 255) & ~(size_t)255;
    return p;
  };
  bf16_t* wqkv_b = (bf16_t*)alloc((size_t)2304 * 768 * 2);
  bf16_t* wprj_b = (bf16_t*)alloc((size_t)768 * 768 * 2);
  bf16_t* fc1w_b = (bf16_t*)alloc((size_t)6144 * 768 * 2);
  bf16_t* fc2w_b = (bf16_t*)alloc((size_t)768 * 3072 * 2);
  bf16_t* h1_b   = (bf16_t*)alloc((size_t)4096 * 768 * 2);
  float*  qkv_f  = (float*) alloc((size_t)4096 * 2304 * 4);
  bf16_t* attn_b = (bf16_t*)alloc((size_t)4096 * 768 * 2);
  float*  x1_f   = (float*) alloc((size_t)4096 * 768 * 4);
  float*  m_f    = (float*) alloc((size_t)BB * NN * 64 * 4);
  float*  nmax_f = (float*) alloc((size_t)BB * 1024 * 4);
  int*    nidx_i = (int*)   alloc((size_t)BB * 1024 * 4);
  int*    ord_i  = (int*)   alloc((size_t)BB * 1024 * 4);
  int*    didx_i = (int*)   alloc((size_t)BB * RR * 4);
  float*  cnt_f  = (float*) alloc((size_t)BB * 1024 * 4);
  float*  dbuf_f = (float*) alloc((size_t)BB * 1024 * 768 * 4);
  float*  xm_f   = (float*) alloc((size_t)BB * NM * 768 * 4);
  bf16_t* h2_b   = (bf16_t*)alloc((size_t)3584 * 768 * 2);
  bf16_t* u_b    = (bf16_t*)alloc((size_t)3584 * 6144 * 2);
  bf16_t* g_b    = (bf16_t*)alloc((size_t)3584 * 3072 * 2);

  // 1) weight conversions to bf16
  {
    int n;
    n = 2304 * 768; cvt_bf16_kernel<<<(n + 255) / 256, 256, 0, stream>>>(w_qkv, wqkv_b, n);
    n = 768 * 768;  cvt_bf16_kernel<<<(n + 255) / 256, 256, 0, stream>>>(w_proj, wprj_b, n);
    n = 6144 * 768; cvt_bf16_kernel<<<(n + 255) / 256, 256, 0, stream>>>(fc1w, fc1w_b, n);
    n = 768 * 3072; cvt_bf16_kernel<<<(n + 255) / 256, 256, 0, stream>>>(fc2w, fc2w_b, n);
  }

  // 2) LN1 -> bf16
  ln_kernel<<<4096, 256, 0, stream>>>(x, n1w, n1b, h1_b);

  // 3) QKV GEMM (4096 x 2304 x 768) -> f32
  gemm_bf16_kernel<<<dim3(2304 / 64, 4096 / 64), 128, 0, stream>>>(
      h1_b, wqkv_b, b_qkv, qkv_f, nullptr, nullptr, nullptr, 4096, 2304, 768);

  // 4) local-window attention -> bf16 (B,N,D)
  attn_kernel<<<(BB * HH * NN) / 4, 128, 0, stream>>>(qkv_f, mask, attn_b);

  // 5) out-proj GEMM + fused residual: x1 = x + gamma_1 * (attn @ Wp^T + bp)
  gemm_bf16_kernel<<<dim3(768 / 64, 4096 / 64), 128, 0, stream>>>(
      attn_b, wprj_b, b_proj, x1_f, nullptr, x, g1, 4096, 768, 768);

  // 6) ToMe metric + sim + argsort + merge
  metric_kernel<<<(BB * NN) / 4, 128, 0, stream>>>(qkv_f, m_f);
  sim_kernel<<<(BB * 1024) / 8, 256, 0, stream>>>(m_f, nmax_f, nidx_i);
  sort_kernel<<<BB, 1024, 0, stream>>>(nmax_f, nidx_i, ord_i, didx_i);
  merge_init_kernel<<<BB * 1024, 256, 0, stream>>>(x1_f, dbuf_f, cnt_f);
  merge_scatter_kernel<<<BB * RR, 256, 0, stream>>>(x1_f, ord_i, didx_i, dbuf_f, cnt_f);
  merge_final_kernel<<<BB * NM, 256, 0, stream>>>(x1_f, ord_i, dbuf_f, cnt_f, xm_f);

  // 7) LN2 -> bf16
  ln_kernel<<<3584, 256, 0, stream>>>(xm_f, n2w, n2b, h2_b);

  // 8) fc1 GEMM (3584 x 6144 x 768) -> bf16 u
  gemm_bf16_kernel<<<dim3(6144 / 64, 3584 / 64), 128, 0, stream>>>(
      h2_b, fc1w_b, fc1b, nullptr, u_b, nullptr, nullptr, 3584, 6144, 768);

  // 9) SwiGLU -> bf16 g
  swiglu_kernel<<<3584, 256, 0, stream>>>(u_b, g_b);

  // 10) fc2 GEMM (3584 x 768 x 3072) + fused final residual:
  //     out = xm + gamma_2 * (g @ Wfc2^T + bfc2)
  gemm_bf16_kernel<<<dim3(768 / 64, 3584 / 64), 128, 0, stream>>>(
      g_b, fc2w_b, fc2b, out, nullptr, xm_f, g2, 3584, 768, 3072);
}
